// DiffusionTransformer_88175678587710
// MI455X (gfx1250) — matrix operand order repaired and checked by reading, not compiled, not run on hardware
//
#include <hip/hip_runtime.h>
#include <hip/hip_bf16.h>

typedef __attribute__((ext_vector_type(16))) _Float16 v16h;
typedef __attribute__((ext_vector_type(8)))  float    v8f;

#define DEVI __device__ __forceinline__

constexpr int Bb   = 2;
constexpr int Nn   = 2048;
constexpr int ACH  = 256;
constexpr int SCH  = 384;
constexpr int ZCH  = 16;
constexpr int Hh   = 8;
constexpr int NQ   = 32;
constexpr int NK   = 128;
constexpr int DH   = 32;
constexpr int HID  = 512;
constexpr int Cc   = 64;
constexpr int Mtok = Bb * Nn;   // 4096 tokens

DEVI float sigf(float x) { return 1.0f / (1.0f + __expf(-x)); }

// ---- WMMA tile loaders -------------------------------------------------
// A-tile (16x32 f16): lanes 0-15 -> row=lane, K chunks [0..7],[16..23];
//                     lanes 16-31 -> row=lane-16, K chunks [8..15],[24..31].
DEVI v16h load_tile_a(const _Float16* p0, int ld, int lane) {
  const _Float16* p = p0 + (size_t)(lane & 15) * ld + ((lane & 16) ? 8 : 0);
  union { float4 f[2]; v16h h; } u;
  u.f[0] = *reinterpret_cast<const float4*>(p);
  u.f[1] = *reinterpret_cast<const float4*>(p + 16);
  return u.h;
}
// B-tile (32x16 f16) loaded from W^T (row n = B column, contiguous K):
// lanes 0-15 -> col=lane, K chunks [0..7],[16..23]; lanes 16-31 -> col=lane-16, K chunks [8..15],[24..31].
DEVI v16h load_tile_b(const _Float16* p0, int ld, int lane) {
  const _Float16* p = p0 + (size_t)(lane & 15) * ld + ((lane & 16) ? 8 : 0);
  union { float4 f[2]; v16h h; } u;
  u.f[0] = *reinterpret_cast<const float4*>(p);
  u.f[1] = *reinterpret_cast<const float4*>(p + 16);
  return u.h;
}

// ---- weight transpose f32 (KxN) -> f16 (NxK) ---------------------------
__global__ __launch_bounds__(256) void k_transpose(const float* W, _Float16* WT, int K, int N) {
  int idx = blockIdx.x * 256 + threadIdx.x;
  if (idx >= K * N) return;
  int k = idx / N, n = idx - k * N;
  WT[(size_t)n * K + k] = (_Float16)W[idx];
}

// ---- token LayerNorms --------------------------------------------------
DEVI float blockSum(float v, float* red) {
  int t = threadIdx.x;
  red[t] = v; __syncthreads();
  for (int o = 128; o > 0; o >>= 1) { if (t < o) red[t] += red[t + o]; __syncthreads(); }
  float r = red[0]; __syncthreads();
  return r;
}

__global__ __launch_bounds__(256) void k_ln_tokens(const float* a, const float* s,
    const float* sw1, const float* sw2,
    float* anf, _Float16* sn1h, _Float16* sn2h, _Float16* sh) {
  __shared__ float red[256];
  int tok = blockIdx.x, t = threadIdx.x;
  float x  = a[(size_t)tok * ACH + t];
  float su = blockSum(x, red);
  float sq = blockSum(x * x, red);
  float mu = su / ACH;
  float var = sq / ACH - mu * mu;
  anf[(size_t)tok * ACH + t] = (x - mu) * rsqrtf(var + 1e-5f);

  const float* sp = s + (size_t)tok * SCH;
  float s0 = sp[t];
  float s1 = (t < SCH - 256) ? sp[256 + t] : 0.0f;
  float ssu = blockSum(s0 + s1, red);
  float ssq = blockSum(s0 * s0 + s1 * s1, red);
  float smu = ssu / SCH;
  float rs  = rsqrtf(ssq / SCH - smu * smu + 1e-5f);
  float n0 = (s0 - smu) * rs;
  sn1h[(size_t)tok * SCH + t] = (_Float16)(n0 * sw1[t]);
  sn2h[(size_t)tok * SCH + t] = (_Float16)(n0 * sw2[t]);
  sh  [(size_t)tok * SCH + t] = (_Float16)s0;
  if (t < SCH - 256) {
    float n1 = (s1 - smu) * rs;
    sn1h[(size_t)tok * SCH + 256 + t] = (_Float16)(n1 * sw1[256 + t]);
    sn2h[(size_t)tok * SCH + 256 + t] = (_Float16)(n1 * sw2[256 + t]);
    sh  [(size_t)tok * SCH + 256 + t] = (_Float16)s1;
  }
}

// ---- WMMA GEMM: Y[M,N] = X[M,K] @ W (W^T given, N x K f16) -------------
// One wave computes a 32x64 output strip: 2 A-tiles x 4 B-tiles = 8 WMMAs
// per 32-wide k-step; each A tile is reused 4x, each B tile 2x.
__global__ __launch_bounds__(256) void k_gemm(const _Float16* X, const _Float16* WT,
    const float* bias, float* Yf, _Float16* Yh, int M, int K, int N) {
  int lane = threadIdx.x & 31;
  int wave = threadIdx.x >> 5;
  int gw = blockIdx.x * 8 + wave;
  int tilesN = N >> 6;                  // 64-wide strips
  int tM = gw / tilesN;
  int tN = gw - tM * tilesN;
  if (tM * 32 >= M) return;
  const _Float16* xb = X  + (size_t)tM * 32 * K;
  const _Float16* wb = WT + (size_t)tN * 64 * K;
  v8f acc[2][4] = {};
  for (int k0 = 0; k0 < K; k0 += 32) {
    v16h a0 = load_tile_a(xb + k0, K, lane);
    v16h a1 = load_tile_a(xb + (size_t)16 * K + k0, K, lane);
    v16h b0 = load_tile_b(wb + k0, K, lane);
    v16h b1 = load_tile_b(wb + (size_t)16 * K + k0, K, lane);
    v16h b2 = load_tile_b(wb + (size_t)32 * K + k0, K, lane);
    v16h b3 = load_tile_b(wb + (size_t)48 * K + k0, K, lane);
    acc[0][0] = __builtin_amdgcn_wmma_f32_16x16x32_f16(false, a0, false, b0, (short)0, acc[0][0], false, false);
    acc[0][1] = __builtin_amdgcn_wmma_f32_16x16x32_f16(false, a0, false, b1, (short)0, acc[0][1], false, false);
    acc[0][2] = __builtin_amdgcn_wmma_f32_16x16x32_f16(false, a0, false, b2, (short)0, acc[0][2], false, false);
    acc[0][3] = __builtin_amdgcn_wmma_f32_16x16x32_f16(false, a0, false, b3, (short)0, acc[0][3], false, false);
    acc[1][0] = __builtin_amdgcn_wmma_f32_16x16x32_f16(false, a1, false, b0, (short)0, acc[1][0], false, false);
    acc[1][1] = __builtin_amdgcn_wmma_f32_16x16x32_f16(false, a1, false, b1, (short)0, acc[1][1], false, false);
    acc[1][2] = __builtin_amdgcn_wmma_f32_16x16x32_f16(false, a1, false, b2, (short)0, acc[1][2], false, false);
    acc[1][3] = __builtin_amdgcn_wmma_f32_16x16x32_f16(false, a1, false, b3, (short)0, acc[1][3], false, false);
  }
  int moff = (lane & 16) ? 8 : 0;
  int nl   = lane & 15;
  for (int mi = 0; mi < 2; ++mi) {
    int mb = tM * 32 + mi * 16 + moff;
    for (int nj = 0; nj < 4; ++nj) {
      int n = tN * 64 + nj * 16 + nl;
      float bv = bias ? bias[n] : 0.0f;
      if (Yh) {
        for (int r = 0; r < 8; ++r)
          Yh[(size_t)(mb + r) * N + n] = (_Float16)(acc[mi][nj][r] + bv);
      } else {
        for (int r = 0; r < 8; ++r)
          Yf[(size_t)(mb + r) * N + n] = acc[mi][nj][r] + bv;
      }
    }
  }
}

// ---- pair-bias: gather z slice, LN, dot with wb (16x8) -----------------
__global__ __launch_bounds__(256) void k_bias(const float* z, const int* key_idx,
    const float* zw, const float* zb, const float* wbm, float* biasT) {
  int idx = blockIdx.x * 256 + threadIdx.x;          // Bb*Cc*NQ*NK = 524288
  int j = idx & (NK - 1);
  int q = (idx >> 7) & (NQ - 1);
  int c = (idx >> 12) & (Cc - 1);
  int b = idx >> 18;
  if (b >= Bb) return;
  int tokq = c * NQ + q;
  int tokk = key_idx[c * NK + j];
  const float* zp = z + (((size_t)b * Nn + tokq) * Nn + tokk) * ZCH;
  float v[ZCH]; float su = 0.f, sq = 0.f;
  for (int i = 0; i < ZCH; ++i) { float vv = zp[i]; v[i] = vv; su += vv; sq += vv * vv; }
  float mu = su / ZCH;
  float rs = rsqrtf(sq / ZCH - mu * mu + 1e-5f);
  for (int i = 0; i < ZCH; ++i) v[i] = (v[i] - mu) * rs * zw[i] + zb[i];
  float* bp = biasT + ((size_t)(b * Cc + c) * Hh) * (NQ * NK) + q * NK + j;
  for (int h = 0; h < Hh; ++h) {
    float acc = 0.f;
    for (int i = 0; i < ZCH; ++i) acc += v[i] * wbm[i * Hh + h];
    bp[(size_t)h * NQ * NK] = acc;
  }
}

// ---- windowed attention, one (b, window, head) per block ---------------
__global__ __launch_bounds__(256) void k_attn(const _Float16* qh, const _Float16* kh,
    const _Float16* vh, const int* key_idx, const int* key_mask,
    const float* biasT, float* ow) {
  int bid = blockIdx.x;                 // Bb*Cc*Hh = 1024
  int h = bid & (Hh - 1);
  int c = (bid >> 3) & (Cc - 1);
  int b = bid >> 9;
  int tid = threadIdx.x;
  int lane = tid & 31, wave = tid >> 5;

  __shared__ __align__(16) _Float16 qs[NQ * DH];
  __shared__ __align__(16) _Float16 ks[NK * DH];
  __shared__ __align__(16) _Float16 vts[DH * NK];
  __shared__ __align__(16) float    lg[NQ * NK];
  __shared__ __align__(16) _Float16 ah[NQ * NK];
  __shared__ int kmk[NK];

  for (int e = tid; e < NQ * DH; e += 256) {
    int qq = e >> 5, d = e & 31;
    qs[e] = qh[((size_t)b * Nn + c * NQ + qq) * ACH + h * DH + d];
  }
  for (int e = tid; e < NK * DH; e += 256) {
    int j = e >> 5, d = e & 31;
    int row = key_idx[c * NK + j];
    size_t src = ((size_t)b * Nn + row) * ACH + h * DH + d;
    ks[e] = kh[src];
    vts[d * NK + j] = vh[src];
  }
  if (tid < NK) kmk[tid] = key_mask[c * NK + tid];
  __syncthreads();

  // logits = q k^T / sqrt(DH) + bias, masked
  const float* bp = biasT + ((size_t)(b * Cc + c) * Hh + h) * (NQ * NK);
  const float scale = 0.1767766952966369f;  // 1/sqrt(32)
  for (int t2 = 0; t2 < 2; ++t2) {
    int tt = wave + t2 * 8;
    int tm = tt >> 3, tk = tt & 7;
    v16h aa = load_tile_a(&qs[tm * 16 * DH], DH, lane);
    v16h bb = load_tile_b(&ks[tk * 16 * DH], DH, lane);
    v8f acc = {};
    acc = __builtin_amdgcn_wmma_f32_16x16x32_f16(false, aa, false, bb, (short)0, acc, false, false);
    int nn = lane & 15;
    int moff = (lane & 16) ? 8 : 0;
    int j = tk * 16 + nn;
    bool mk = kmk[j] != 0;
    for (int r = 0; r < 8; ++r) {
      int m = tm * 16 + moff + r;
      float val = acc[r] * scale + bp[m * NK + j];
      lg[m * NK + j] = mk ? val : -1.0e9f;
    }
  }
  __syncthreads();

  // softmax over 128 keys: 8-lane team per row
  {
    int row = tid >> 3, i8 = tid & 7;
    float vals[16]; float mx = -3.0e38f;
    for (int p = 0; p < 16; ++p) {
      float vv = lg[row * NK + i8 * 16 + p];
      vals[p] = vv; mx = fmaxf(mx, vv);
    }
    for (int off = 1; off < 8; off <<= 1) mx = fmaxf(mx, __shfl_xor(mx, off, 8));
    float sum = 0.f;
    for (int p = 0; p < 16; ++p) { vals[p] = __expf(vals[p] - mx); sum += vals[p]; }
    for (int off = 1; off < 8; off <<= 1) sum += __shfl_xor(sum, off, 8);
    float inv = 1.0f / sum;
    for (int p = 0; p < 16; ++p) ah[row * NK + i8 * 16 + p] = (_Float16)(vals[p] * inv);
  }
  __syncthreads();

  // out = attn(32x128) @ v(128x32)
  if (wave < 4) {
    int tm = wave >> 1, tn = wave & 1;
    v8f acc = {};
    for (int kk = 0; kk < 4; ++kk) {
      v16h aa = load_tile_a(&ah[tm * 16 * NK + kk * 32], NK, lane);
      v16h bb = load_tile_b(&vts[tn * 16 * NK + kk * 32], NK, lane);
      acc = __builtin_amdgcn_wmma_f32_16x16x32_f16(false, aa, false, bb, (short)0, acc, false, false);
    }
    int nn = lane & 15;
    int moff = (lane & 16) ? 8 : 0;
    for (int r = 0; r < 8; ++r) {
      int m = tm * 16 + moff + r;
      ow[((size_t)b * Nn + c * NQ + m) * ACH + h * DH + tn * 16 + nn] = acc[r];
    }
  }
}

// ---- elementwise fusions ----------------------------------------------
__global__ __launch_bounds__(256) void k_adaln_mix(const float* G1, const float* G2,
    const float* b1, const float* anf, _Float16* outh, int total) {
  int idx = blockIdx.x * 256 + threadIdx.x;
  if (idx >= total) return;
  int n = idx & (ACH - 1);
  outh[idx] = (_Float16)(sigf(G1[idx] + b1[n]) * anf[idx] + G2[idx]);
}
__global__ __launch_bounds__(256) void k_gate(const float* G, const float* bg,
    const float* x, _Float16* outh, int total) {
  int idx = blockIdx.x * 256 + threadIdx.x;
  if (idx >= total) return;
  int n = idx & (ACH - 1);
  outh[idx] = (_Float16)(sigf(G[idx] + bg[n]) * x[idx]);
}
__global__ __launch_bounds__(256) void k_swiglu(const float* H1, const float* H2,
    _Float16* outh, int total) {
  int idx = blockIdx.x * 256 + threadIdx.x;
  if (idx >= total) return;
  float x = H1[idx];
  outh[idx] = (_Float16)(x * sigf(x) * H2[idx]);
}
__global__ __launch_bounds__(256) void k_final(const float* bout, const float* G6,
    const float* bg2, const float* Tt, float* out, int total) {
  int idx = blockIdx.x * 256 + threadIdx.x;
  if (idx >= total) return;
  int n = idx & (ACH - 1);
  out[idx] = bout[idx] + sigf(G6[idx] + bg2[n]) * Tt[idx];
}

// ------------------------------------------------------------------------
extern "C" void kernel_launch(void* const* d_in, const int* in_sizes, int n_in,
                              void* d_out, int out_size, void* d_ws, size_t ws_size,
                              hipStream_t stream) {
  (void)in_sizes; (void)n_in; (void)out_size; (void)ws_size;
  const float* a        = (const float*)d_in[0];
  const float* s        = (const float*)d_in[1];
  const float* z        = (const float*)d_in[2];
  const float* ada1_s_w = (const float*)d_in[3];
  const float* ada1_l1w = (const float*)d_in[4];
  const float* ada1_l1b = (const float*)d_in[5];
  const float* ada1_l2w = (const float*)d_in[6];
  const float* wq       = (const float*)d_in[7];
  const float* bq       = (const float*)d_in[8];
  const float* wk       = (const float*)d_in[9];
  const float* wv       = (const float*)d_in[10];
  const float* z_ln_w   = (const float*)d_in[11];
  const float* z_ln_b   = (const float*)d_in[12];
  const float* wbm      = (const float*)d_in[13];
  const float* wg       = (const float*)d_in[14];
  const float* bg       = (const float*)d_in[15];
  const float* wo       = (const float*)d_in[16];
  const float* ada2_s_w = (const float*)d_in[17];
  const float* ada2_l1w = (const float*)d_in[18];
  const float* ada2_l1b = (const float*)d_in[19];
  const float* ada2_l2w = (const float*)d_in[20];
  const float* w1       = (const float*)d_in[21];
  const float* w2       = (const float*)d_in[22];
  const float* w3       = (const float*)d_in[23];
  const float* wg2      = (const float*)d_in[24];
  const float* bg2      = (const float*)d_in[25];
  const int*   key_idx  = (const int*)d_in[26];
  const int*   key_mask = (const int*)d_in[27];
  float* out = (float*)d_out;

  uintptr_t cur = (uintptr_t)d_ws;
  auto ALLOC = [&](size_t bytes) -> void* {
    uintptr_t p = (cur + 255) & ~(uintptr_t)255;
    cur = p + bytes;
    return (void*)p;
  };
  // f16 transposed weights
  _Float16* wtL1   = (_Float16*)ALLOC((size_t)ACH * SCH * 2);
  _Float16* wtL2   = (_Float16*)ALLOC((size_t)ACH * SCH * 2);
  _Float16* wtQ    = (_Float16*)ALLOC((size_t)ACH * ACH * 2);
  _Float16* wtK    = (_Float16*)ALLOC((size_t)ACH * ACH * 2);
  _Float16* wtV    = (_Float16*)ALLOC((size_t)ACH * ACH * 2);
  _Float16* wtG    = (_Float16*)ALLOC((size_t)ACH * SCH * 2);
  _Float16* wtO    = (_Float16*)ALLOC((size_t)ACH * ACH * 2);
  _Float16* wtA2L1 = (_Float16*)ALLOC((size_t)ACH * SCH * 2);
  _Float16* wtA2L2 = (_Float16*)ALLOC((size_t)ACH * SCH * 2);
  _Float16* wtW1   = (_Float16*)ALLOC((size_t)HID * ACH * 2);
  _Float16* wtW2   = (_Float16*)ALLOC((size_t)HID * ACH * 2);
  _Float16* wtW3   = (_Float16*)ALLOC((size_t)ACH * HID * 2);
  _Float16* wtG2   = (_Float16*)ALLOC((size_t)ACH * SCH * 2);
  // activations
  float*    anf   = (float*)   ALLOC((size_t)Mtok * ACH * 4);
  _Float16* sn1h  = (_Float16*)ALLOC((size_t)Mtok * SCH * 2);
  _Float16* sn2h  = (_Float16*)ALLOC((size_t)Mtok * SCH * 2);
  _Float16* sh    = (_Float16*)ALLOC((size_t)Mtok * SCH * 2);
  float*    P1    = (float*)   ALLOC((size_t)Mtok * ACH * 4);
  float*    P2    = (float*)   ALLOC((size_t)Mtok * ACH * 4);
  _Float16* an1h  = (_Float16*)ALLOC((size_t)Mtok * ACH * 2);
  _Float16* an2h  = (_Float16*)ALLOC((size_t)Mtok * ACH * 2);
  _Float16* qhB   = (_Float16*)ALLOC((size_t)Mtok * ACH * 2);
  _Float16* khB   = (_Float16*)ALLOC((size_t)Mtok * ACH * 2);
  _Float16* vhB   = (_Float16*)ALLOC((size_t)Mtok * ACH * 2);
  float*    biasT = (float*)   ALLOC((size_t)Bb * Cc * Hh * NQ * NK * 4);
  float*    owB   = (float*)   ALLOC((size_t)Mtok * ACH * 4);
  _Float16* goh   = (_Float16*)ALLOC((size_t)Mtok * ACH * 2);
  float*    bout  = (float*)   ALLOC((size_t)Mtok * ACH * 4);
  float*    H1    = (float*)   ALLOC((size_t)Mtok * HID * 4);
  float*    H2    = (float*)   ALLOC((size_t)Mtok * HID * 4);
  _Float16* hidh  = (_Float16*)ALLOC((size_t)Mtok * HID * 2);
  float*    Tt    = (float*)   ALLOC((size_t)Mtok * ACH * 4);

  auto TP = [&](const float* W, _Float16* WT, int K, int N) {
    int total = K * N;
    k_transpose<<<(total + 255) / 256, 256, 0, stream>>>(W, WT, K, N);
  };
  auto GEMM = [&](const _Float16* X, const _Float16* WT, const float* bias,
                  float* Yf, _Float16* Yh, int M, int K, int N) {
    int waves = (M / 32) * (N / 64);
    k_gemm<<<(waves + 7) / 8, 256, 0, stream>>>(X, WT, bias, Yf, Yh, M, K, N);
  };

  TP(ada1_l1w, wtL1, SCH, ACH);   TP(ada1_l2w, wtL2, SCH, ACH);
  TP(wq, wtQ, ACH, ACH);          TP(wk, wtK, ACH, ACH);        TP(wv, wtV, ACH, ACH);
  TP(wg, wtG, SCH, ACH);          TP(wo, wtO, ACH, ACH);
  TP(ada2_l1w, wtA2L1, SCH, ACH); TP(ada2_l2w, wtA2L2, SCH, ACH);
  TP(w1, wtW1, ACH, HID);         TP(w2, wtW2, ACH, HID);       TP(w3, wtW3, HID, ACH);
  TP(wg2, wtG2, SCH, ACH);

  k_ln_tokens<<<Mtok, 256, 0, stream>>>(a, s, ada1_s_w, ada2_s_w, anf, sn1h, sn2h, sh);

  // adaLN 1 -> an1
  GEMM(sn1h, wtL1, nullptr, P1, nullptr, Mtok, SCH, ACH);
  GEMM(sn1h, wtL2, nullptr, P2, nullptr, Mtok, SCH, ACH);
  k_adaln_mix<<<(Mtok * ACH + 255) / 256, 256, 0, stream>>>(P1, P2, ada1_l1b, anf, an1h, Mtok * ACH);

  // QKV
  GEMM(an1h, wtQ, bq,      nullptr, qhB, Mtok, ACH, ACH);
  GEMM(an1h, wtK, nullptr, nullptr, khB, Mtok, ACH, ACH);
  GEMM(an1h, wtV, nullptr, nullptr, vhB, Mtok, ACH, ACH);

  // pair bias + attention
  k_bias<<<(Bb * Cc * NQ * NK) / 256, 256, 0, stream>>>(z, key_idx, z_ln_w, z_ln_b, wbm, biasT);
  k_attn<<<Bb * Cc * Hh, 256, 0, stream>>>(qhB, khB, vhB, key_idx, key_mask, biasT, owB);

  // gate + output projection
  GEMM(sh, wtG, nullptr, P1, nullptr, Mtok, SCH, ACH);
  k_gate<<<(Mtok * ACH + 255) / 256, 256, 0, stream>>>(P1, bg, owB, goh, Mtok * ACH);
  GEMM(goh, wtO, nullptr, bout, nullptr, Mtok, ACH, ACH);

  // adaLN 2 -> an2
  GEMM(sn2h, wtA2L1, nullptr, P1, nullptr, Mtok, SCH, ACH);
  GEMM(sn2h, wtA2L2, nullptr, P2, nullptr, Mtok, SCH, ACH);
  k_adaln_mix<<<(Mtok * ACH + 255) / 256, 256, 0, stream>>>(P1, P2, ada2_l1b, anf, an2h, Mtok * ACH);

  // SwiGLU transition
  GEMM(an2h, wtW1, nullptr, H1, nullptr, Mtok, ACH, HID);
  GEMM(an2h, wtW2, nullptr, H2, nullptr, Mtok, ACH, HID);
  k_swiglu<<<(Mtok * HID + 255) / 256, 256, 0, stream>>>(H1, H2, hidh, Mtok * HID);
  GEMM(hidh, wtW3, nullptr, Tt, nullptr, Mtok, HID, ACH);

  // final gate + sum
  GEMM(sh, wtG2, nullptr, P1, nullptr, Mtok, SCH, ACH);
  k_final<<<(Mtok * ACH + 255) / 256, 256, 0, stream>>>(bout, P1, bg2, Tt, out, Mtok * ACH);
}
